// LiveNet_77429670412802
// MI455X (gfx1250) — matrix operand order repaired and checked by reading, not compiled, not run on hardware
//
#include <hip/hip_runtime.h>

// ---- CDNA5 WMMA vector types (wave32) ----
typedef __attribute__((ext_vector_type(16))) __bf16 v16bf;
typedef __attribute__((ext_vector_type(8)))  __bf16 v8bf;
typedef __attribute__((ext_vector_type(8)))  float  v8f;
typedef __attribute__((ext_vector_type(4)))  float  f4;

#define N_IN   64
#define N_MID  128
#define N_OUT  32
#define NWAVES 8
// LDS row pads: stride must be a multiple of 8 bf16 (16B align for b128 loads)
// and give distinct bank groups for 16 consecutive rows.
#define W1_PAD 88    // 176 B = 44 dwords: 44*n mod 64 distinct for n=0..15
#define W2_PAD 136   // 272 B = 68 dwords: 4*n mod 64 distinct
#define H_PAD  136

__device__ __forceinline__ v16bf cat8(v8bf lo, v8bf hi) {
    return __builtin_shufflevector(lo, hi, 0,1,2,3,4,5,6,7,8,9,10,11,12,13,14,15);
}

__global__ __launch_bounds__(256) void livenet_mlp_kernel(
    const float* __restrict__ x,  const float* __restrict__ W1,
    const float* __restrict__ b1, const float* __restrict__ W2,
    const float* __restrict__ b2, float* __restrict__ y, int nTiles)
{
    // bf16 weights, transposed [n][k] so B-fragments are contiguous LDS reads
    __shared__ __bf16 sW1T[N_MID][W1_PAD];   // 22.0 KB
    __shared__ __bf16 sW2T[N_OUT][W2_PAD];   //  8.5 KB
    __shared__ float  sB1[N_MID];
    __shared__ float  sB2[N_OUT];
    __shared__ __bf16 sH[NWAVES][16][H_PAD]; // 34.0 KB per-wave hidden scratch

    const int tid = threadIdx.x;

    // ---- one-time per-block weight conversion (48 KB, L2-resident) ----
    for (int i = tid; i < N_IN * N_MID; i += 256) {      // W1 is [k][n], n fastest
        int k = i >> 7, n = i & 127;
        sW1T[n][k] = (__bf16)W1[i];
    }
    for (int i = tid; i < N_MID * N_OUT; i += 256) {     // W2 is [k][n]
        int k = i >> 5, n = i & 31;
        sW2T[n][k] = (__bf16)W2[i];
    }
    for (int i = tid; i < N_MID; i += 256) sB1[i] = b1[i];
    if (tid < N_OUT) sB2[tid] = b2[tid];
    __syncthreads();

    const int wave  = tid >> 5;
    const int lane  = tid & 31;
    const int lhalf = lane >> 4;   // which half-wave
    const int l16   = lane & 15;

    for (int t = blockIdx.x * NWAVES + wave; t < nTiles; t += gridDim.x * NWAVES) {
        const long row0 = (long)t * 16;

        // ---- layer-1 A fragments: x[16x64] bf16, 2 K-chunks of 32 ----
        // A layout (ISA 7.12.2): lane half 0 -> K {kc*32+0..7, +16..23},
        //                        lane half 1 -> K {kc*32+8..15, +24..31}
        v16bf a1[2];
        #pragma unroll
        for (int kc = 0; kc < 2; ++kc) {
            const int kbase = kc * 32 + lhalf * 8;
            const float* xr = x + (row0 + l16) * N_IN + kbase;
            f4 f0 = __builtin_nontemporal_load((const f4*)(xr));
            f4 f1 = __builtin_nontemporal_load((const f4*)(xr + 4));
            f4 f2 = __builtin_nontemporal_load((const f4*)(xr + 16));
            f4 f3 = __builtin_nontemporal_load((const f4*)(xr + 20));
            v16bf a;
            #pragma unroll
            for (int j = 0; j < 4; ++j) {
                a[j]      = (__bf16)f0[j];
                a[j + 4]  = (__bf16)f1[j];
                a[j + 8]  = (__bf16)f2[j];
                a[j + 12] = (__bf16)f3[j];
            }
            a1[kc] = a;
        }

        // ---- layer 1: H = relu(x@W1 + b1), 8 tiles of 16 cols ----
        #pragma unroll
        for (int nt = 0; nt < 8; ++nt) {
            v8f acc = {};
            #pragma unroll
            for (int kc = 0; kc < 2; ++kc) {
                // B layout: lane = column N; lanes 0-15 hold K kc*32+{0..7,16..23},
                // lanes 16-31 hold K kc*32+{8..15,24..31} (matches A, 2x b128)
                const __bf16* wp = &sW1T[nt * 16 + l16][kc * 32 + lhalf * 8];
                v16bf bf = cat8(*(const v8bf*)wp, *(const v8bf*)(wp + 16));
                acc = __builtin_amdgcn_wmma_f32_16x16x32_bf16(
                        false, a1[kc], false, bf, (short)0, acc, false, false);
            }
            const float bias = sB1[nt * 16 + l16];
            // C layout: VGPR r, lane -> (M = r + lhalf*8, N = l16); bias+ReLU,
            // scatter bf16 into per-wave H scratch (same-wave DS ordering).
            #pragma unroll
            for (int r = 0; r < 8; ++r) {
                float h = acc[r] + bias;
                h = h > 0.f ? h : 0.f;
                sH[wave][r + lhalf * 8][nt * 16 + l16] = (__bf16)h;
            }
        }

        // ---- layer-2 A fragments from H[16x128], 4 K-chunks ----
        v16bf ah[4];
        #pragma unroll
        for (int kc = 0; kc < 4; ++kc) {
            const int kbase = kc * 32 + lhalf * 8;
            const __bf16* hp = &sH[wave][l16][kbase];
            ah[kc] = cat8(*(const v8bf*)hp, *(const v8bf*)(hp + 16));
        }

        // ---- layer 2: y = H@W2 + b2, 2 tiles of 16 cols ----
        float* yr = y + row0 * N_OUT;
        #pragma unroll
        for (int ot = 0; ot < 2; ++ot) {
            v8f acc = {};
            #pragma unroll
            for (int kc = 0; kc < 4; ++kc) {
                const __bf16* wp = &sW2T[ot * 16 + l16][kc * 32 + lhalf * 8];
                v16bf bf = cat8(*(const v8bf*)wp, *(const v8bf*)(wp + 16));
                acc = __builtin_amdgcn_wmma_f32_16x16x32_bf16(
                        false, ah[kc], false, bf, (short)0, acc, false, false);
            }
            const float bias = sB2[ot * 16 + l16];
            #pragma unroll
            for (int r = 0; r < 8; ++r) {
                __builtin_nontemporal_store(acc[r] + bias,
                    yr + (long)(r + lhalf * 8) * N_OUT + ot * 16 + l16);
            }
        }
    }
}

extern "C" void kernel_launch(void* const* d_in, const int* in_sizes, int n_in,
                              void* d_out, int out_size, void* d_ws, size_t ws_size,
                              hipStream_t stream) {
    const float* x  = (const float*)d_in[0];
    const float* W1 = (const float*)d_in[1];
    const float* b1 = (const float*)d_in[2];
    const float* W2 = (const float*)d_in[3];
    const float* b2 = (const float*)d_in[4];
    float* y = (float*)d_out;

    const int B = in_sizes[0] / N_IN;      // 262144
    const int nTiles = B / 16;             // 16384 (B is a multiple of 16)
    int blocks = (nTiles + NWAVES - 1) / NWAVES;
    if (blocks > 1024) blocks = 1024;      // grid-stride: 2 tiles/wave at full size

    livenet_mlp_kernel<<<blocks, 256, 0, stream>>>(x, W1, b1, W2, b2, y, nTiles);
}
